// DiffusionModel_19413252178157
// MI455X (gfx1250) — matrix operand order repaired and checked by reading, not compiled, not run on hardware
//
#include <hip/hip_runtime.h>
#include <hip/hip_bf16.h>

// GNN 2-layer: out = mean-scatter(h@W2^T+b2), h = relu(mean-scatter(x@W1^T+b1))
// Optimized via linearity: aggregate-then-transform. Scatter = L2-atomic bound,
// GEMM = v_wmma_f32_16x16x32_f16 with split-f16 (3-product) fp32-accuracy trick.
// Weights are pre-split to f16 hi/lo ONCE so the WMMA loop is load+MMA only.

typedef __attribute__((ext_vector_type(16))) _Float16 v16h;
typedef __attribute__((ext_vector_type(8)))  float    v8f;

#define C_DIM 128

// ---------------------------------------------------------------- zero
__global__ __launch_bounds__(256) void zero4_kernel(float4* __restrict__ p, int n4) {
  int i = blockIdx.x * blockDim.x + threadIdx.x;
  const int stride = gridDim.x * blockDim.x;
  const float4 z = make_float4(0.f, 0.f, 0.f, 0.f);
  for (; i < n4; i += stride) p[i] = z;
}

// ---------------------------------------------------------------- weight split
// W (f32) -> Wh + Wl (f16 hi part + f16 residual), row-major, done once.
__global__ __launch_bounds__(256) void weight_split_kernel(
    const float* __restrict__ W, _Float16* __restrict__ Wh,
    _Float16* __restrict__ Wl, int n)
{
  int i = blockIdx.x * blockDim.x + threadIdx.x;
  const int stride = gridDim.x * blockDim.x;
  for (; i < n; i += stride) {
    const float w = W[i];
    const _Float16 h = (_Float16)w;
    Wh[i] = h;
    Wl[i] = (_Float16)(w - (float)h);
  }
}

// ---------------------------------------------------------------- scatter-add
// One wave per edge (grid-stride): lane l handles channels [4l, 4l+4).
__global__ __launch_bounds__(256) void scatter_add_kernel(
    const float* __restrict__ feat, const int* __restrict__ src,
    const int* __restrict__ dst, float* __restrict__ agg,
    float* __restrict__ cnt, int E, int nodes)
{
  const int gtid = blockIdx.x * blockDim.x + threadIdx.x;
  const int wid  = gtid >> 5;
  const int lane = gtid & 31;
  const int nw   = (gridDim.x * blockDim.x) >> 5;
  for (int e = wid; e < E; e += nw) {
    const int s = src[e];
    const int d = dst[e];
    if ((unsigned)s >= (unsigned)nodes || (unsigned)d >= (unsigned)nodes) continue;
    // prefetch the feature row for this wave's next edge (global_prefetch_b8)
    const int epf = e + nw;
    if (epf < E) {
      const int spf = src[epf];
      if ((unsigned)spf < (unsigned)nodes)
        __builtin_prefetch(feat + (size_t)spf * C_DIM + lane * 4, 0, 0);
    }
    const float4 v = *(const float4*)(feat + (size_t)s * C_DIM + lane * 4);
    float* o = agg + (size_t)d * C_DIM + lane * 4;
    atomicAdd(o + 0, v.x);
    atomicAdd(o + 1, v.y);
    atomicAdd(o + 2, v.z);
    atomicAdd(o + 3, v.w);
    if (cnt != nullptr && lane == 0) atomicAdd(cnt + d, 1.0f);
  }
}

// ---------------------------------------------------------------- WMMA GEMM
__device__ __forceinline__ void split_to_f16(const float* f, v16h& hi, v16h& lo) {
#pragma unroll
  for (int i = 0; i < 16; ++i) {
    const float v = f[i];
    const _Float16 h = (_Float16)v;
    hi[i] = h;
    lo[i] = (_Float16)(v - (float)h);
  }
}

// Out[m, n] = act( (A[m,:]/max(cnt[m],1)) @ W[n,:] + bias[n]*[cnt[m]>0] )
// One wave per 16-row M-tile, N = 128 (8 tiles), K = 128 (4 steps of 32).
// Weights arrive pre-split as f16 hi/lo (row-major [128][128]).
template <bool RELU>
__global__ __launch_bounds__(256) void gnn_gemm_kernel(
    const float* __restrict__ A, const float* __restrict__ cnt,
    const _Float16* __restrict__ Wh, const _Float16* __restrict__ Wl,
    const float* __restrict__ bias, float* __restrict__ Out, int mtiles)
{
  const int wave = blockIdx.x * (blockDim.x >> 5) + (threadIdx.x >> 5);
  if (wave >= mtiles) return;          // wave-uniform exit: EXEC stays all-ones
  const int lane     = threadIdx.x & 31;
  const int half_sel = lane >> 4;      // 0: lanes 0-15, 1: lanes 16-31
  const int lrow     = lane & 15;
  const int m0       = wave * 16;

  // ---- A fragments for all 4 K-steps (contiguous-k A order used here: M = lane%16,
  //      halfs 0..15 = K kbase+16*half..+15, matching B)
  const int   mA  = m0 + lrow;
  const float inv = 1.0f / fmaxf(cnt[mA], 1.0f);
  const float* __restrict__ arow = A + (size_t)mA * C_DIM;

  v16h Ah[4], Al[4];
#pragma unroll
  for (int ks = 0; ks < 4; ++ks) {
    const int kbase = ks * 32;
    const int kA0 = kbase + half_sel * 16;
    const int kA1 = kbase + half_sel * 16 + 8;
    const float4 a0 = *(const float4*)(arow + kA0);
    const float4 a1 = *(const float4*)(arow + kA0 + 4);
    const float4 a2 = *(const float4*)(arow + kA1);
    const float4 a3 = *(const float4*)(arow + kA1 + 4);
    float af[16];
    af[0]  = a0.x * inv; af[1]  = a0.y * inv; af[2]  = a0.z * inv; af[3]  = a0.w * inv;
    af[4]  = a1.x * inv; af[5]  = a1.y * inv; af[6]  = a1.z * inv; af[7]  = a1.w * inv;
    af[8]  = a2.x * inv; af[9]  = a2.y * inv; af[10] = a2.z * inv; af[11] = a2.w * inv;
    af[12] = a3.x * inv; af[13] = a3.y * inv; af[14] = a3.z * inv; af[15] = a3.w * inv;
    split_to_f16(af, Ah[ks], Al[ks]);
  }

  // ---- epilogue gates (C/D layout: VGPR r -> row m0 + 8*half_sel + r)
  const int crow = m0 + half_sel * 8;
  float gate[8];
#pragma unroll
  for (int r = 0; r < 8; ++r) gate[r] = (cnt[crow + r] > 0.0f) ? 1.0f : 0.0f;

  // ---- per N-tile: 12 WMMAs (3-product split x 4 K-steps), fused epilogue
#pragma unroll
  for (int nt = 0; nt < 8; ++nt) {
    const int n = nt * 16 + lrow;            // B layout: lane = column N
    const _Float16* __restrict__ bhp = Wh + (size_t)n * C_DIM + 16 * half_sel;
    const _Float16* __restrict__ blp = Wl + (size_t)n * C_DIM + 16 * half_sel;

    v8f acc = {};
#pragma unroll
    for (int ks = 0; ks < 4; ++ks) {
      const v16h Bh = *(const v16h*)(bhp + ks * 32);   // 32B-aligned v16h loads
      const v16h Bl = *(const v16h*)(blp + ks * 32);
      acc = __builtin_amdgcn_wmma_f32_16x16x32_f16(
          false, Ah[ks], false, Bh, (short)0, acc, false, false);
      acc = __builtin_amdgcn_wmma_f32_16x16x32_f16(
          false, Al[ks], false, Bh, (short)0, acc, false, false);
      acc = __builtin_amdgcn_wmma_f32_16x16x32_f16(
          false, Ah[ks], false, Bl, (short)0, acc, false, false);
    }

    const float bn = bias[n];
#pragma unroll
    for (int r = 0; r < 8; ++r) {
      float v = acc[r] + bn * gate[r];
      if (RELU) v = fmaxf(v, 0.0f);
      Out[(size_t)(crow + r) * C_DIM + n] = v;
    }
  }
}

// ---------------------------------------------------------------- launch
extern "C" void kernel_launch(void* const* d_in, const int* in_sizes, int n_in,
                              void* d_out, int out_size, void* d_ws, size_t ws_size,
                              hipStream_t stream) {
  const float* x  = (const float*)d_in[0];
  const int*   ei = (const int*)d_in[1];   // int32 per harness convention
  const float* W1 = (const float*)d_in[2];
  const float* b1 = (const float*)d_in[3];
  const float* W2 = (const float*)d_in[4];
  const float* b2 = (const float*)d_in[5];
  float* out = (float*)d_out;

  const int nodes = in_sizes[0] / C_DIM;   // 50000
  const int E     = in_sizes[1] / 2;       // 800000
  const int* srcI = ei;
  const int* dstI = ei + E;

  // workspace: [agg][cnt][h][W1h][W1l][W2h][W2l]  (~51.5 MB)
  float* agg  = (float*)d_ws;
  float* cntp = agg + (size_t)nodes * C_DIM;
  float* h    = cntp + nodes;
  _Float16* W1h = (_Float16*)(h + (size_t)nodes * C_DIM);
  _Float16* W1l = W1h + C_DIM * C_DIM;
  _Float16* W2h = W1l + C_DIM * C_DIM;
  _Float16* W2l = W2h + C_DIM * C_DIM;

  const int featN      = nodes * C_DIM;
  const int mtiles     = nodes / 16;       // 3125 (exact)
  const int gemmBlocks = (mtiles + 7) / 8; // 8 waves (256 thr) per block

  // one-time weight split (tiny: 2 x 16K elements)
  weight_split_kernel<<<64, 256, 0, stream>>>(W1, W1h, W1l, C_DIM * C_DIM);
  weight_split_kernel<<<64, 256, 0, stream>>>(W2, W2h, W2l, C_DIM * C_DIM);

  // ---- layer 1: agg = scatter_sum(x[src] -> dst), cnt = in-degree
  zero4_kernel<<<1024, 256, 0, stream>>>((float4*)agg, (featN + nodes) / 4);
  scatter_add_kernel<<<2048, 256, 0, stream>>>(x, srcI, dstI, agg, cntp, E, nodes);
  gnn_gemm_kernel<true><<<gemmBlocks, 256, 0, stream>>>(agg, cntp, W1h, W1l, b1, h, mtiles);

  // ---- layer 2: reuse agg buffer; cnt is identical (same dst indices)
  zero4_kernel<<<1024, 256, 0, stream>>>((float4*)agg, featN / 4);
  scatter_add_kernel<<<2048, 256, 0, stream>>>(h, srcI, dstI, agg, nullptr, E, nodes);
  gnn_gemm_kernel<false><<<gemmBlocks, 256, 0, stream>>>(agg, cntp, W2h, W2l, b2, out, mtiles);

  (void)n_in; (void)out_size; (void)ws_size;
}
